// SparseMoEFFN_27384711479464
// MI455X (gfx1250) — matrix operand order repaired and checked by reading, not compiled, not run on hardware
//
#include <hip/hip_runtime.h>
#include <math.h>

#define D_MODEL 512
#define D_FF    2048
#define NEXP    8
#define ROWS    32          // tokens per FFN block (N dimension)
#define FCH     64          // d_ff chunk per iteration
#define XB_STRIDE 520       // padded bf16 LDS stride for x tile (bank-conflict-free b128)
#define HB_STRIDE 72        // padded bf16 LDS stride for h tile

typedef float v8f  __attribute__((ext_vector_type(8)));
typedef __bf16 v8bf  __attribute__((ext_vector_type(8)));
typedef __bf16 v16bf __attribute__((ext_vector_type(16)));
typedef unsigned short v8u __attribute__((ext_vector_type(8)));

// ---- fp32 <-> bf16 split helpers (round-to-nearest-even) ----
__device__ __forceinline__ unsigned short f2bf(float f) {
  unsigned u = __float_as_uint(f);
  u += 0x7FFFu + ((u >> 16) & 1u);
  return (unsigned short)(u >> 16);
}
__device__ __forceinline__ float bf2f(unsigned short h) {
  return __uint_as_float((unsigned)h << 16);
}

// D(16x16 f32) += A(16x32 bf16) * B(32x16 bf16), fp32 accumulate
__device__ __forceinline__ v8f wmma_bf(v16bf a, v16bf b, v8f c) {
  return __builtin_amdgcn_wmma_f32_16x16x32_bf16(
      false, a, false, b, (short)0, c, false, false);
}

// A operand per ISA 16-bit layout: lane holds K-runs [p, p+8) and [p+16, p+24)
__device__ __forceinline__ v16bf ldA(const __bf16* p) {
  v8bf a = *(const v8bf*)p;
  v8bf b = *(const v8bf*)(p + 16);
  return __builtin_shufflevector(a, b, 0,1,2,3,4,5,6,7,8,9,10,11,12,13,14,15);
}
// B operand: lane holds K-runs [p, p+8) and [p+16, p+24)
__device__ __forceinline__ v16bf ldB(const __bf16* p) {
  v8bf a = *(const v8bf*)p;
  v8bf b = *(const v8bf*)(p + 16);
  return __builtin_shufflevector(a, b, 0,1,2,3,4,5,6,7,8,9,10,11,12,13,14,15);
}

// jax.nn.gelu default: tanh approximation
__device__ __forceinline__ float gelu_tanh(float v) {
  const float k = 0.7978845608028654f;  // sqrt(2/pi)
  float t = k * (v + 0.044715f * v * v * v);
  return 0.5f * v * (1.0f + tanhf(t));
}

__global__ void moe_zero_kernel(float* __restrict__ out, int* __restrict__ counts, int n) {
  int i = blockIdx.x * blockDim.x + threadIdx.x;
  if (i < NEXP) counts[i] = 0;
  for (int j = i; j < n; j += gridDim.x * blockDim.x) out[j] = 0.0f;
}

// Transpose [E][R][C] f32 -> [E][C][R] bf16 hi/lo split.
__global__ void transpose_split_kernel(const float* __restrict__ in,
                                       unsigned short* __restrict__ outh,
                                       unsigned short* __restrict__ outl,
                                       int R, int C) {
  __shared__ float tile[32][33];
  int e  = blockIdx.z;
  int c0 = blockIdx.x * 32;
  int r0 = blockIdx.y * 32;
  int tc = threadIdx.x & 31;
  int tr = threadIdx.x >> 5;     // 0..7
  const float* inp = in + (size_t)e * R * C;
  #pragma unroll
  for (int i = 0; i < 4; ++i) {
    int r = tr + i * 8;
    tile[r][tc] = inp[(size_t)(r0 + r) * C + c0 + tc];
  }
  __syncthreads();
  size_t obase = (size_t)e * C * R;
  #pragma unroll
  for (int i = 0; i < 4; ++i) {
    int r = tr + i * 8;
    float v = tile[tc][r];                 // = in[r0+tc][c0+r]
    unsigned short h = f2bf(v);
    unsigned short l = f2bf(v - bf2f(h));
    size_t oidx = obase + (size_t)(c0 + r) * R + r0 + tc;
    outh[oidx] = h;
    outl[oidx] = l;
  }
}

// One wave per token: logits, top-2 softmax weights, atomic append to expert buckets.
__global__ void moe_router_kernel(const float* __restrict__ x,
                                  const float* __restrict__ Wr,
                                  const float* __restrict__ br,
                                  int* __restrict__ counts,
                                  int* __restrict__ tids,
                                  float* __restrict__ tws,
                                  int ntok) {
  int wave = threadIdx.x >> 5;
  int lane = threadIdx.x & 31;
  int n = blockIdx.x * (blockDim.x >> 5) + wave;
  if (n >= ntok) return;
  const float* xr = x + (size_t)n * D_MODEL;
  float logit[NEXP];
  for (int e = 0; e < NEXP; ++e) {
    const float* wr = Wr + e * D_MODEL;
    float p = 0.0f;
    for (int d = lane; d < D_MODEL; d += 32) p += xr[d] * wr[d];
    for (int off = 16; off > 0; off >>= 1) p += __shfl_xor(p, off, 32);
    logit[e] = p + br[e];
  }
  if (lane == 0) {
    int i0 = 0; float m0 = logit[0];
    for (int e = 1; e < NEXP; ++e) if (logit[e] > m0) { m0 = logit[e]; i0 = e; }
    int i1 = -1; float m1 = -3.4e38f;
    for (int e = 0; e < NEXP; ++e) if (e != i0 && logit[e] > m1) { m1 = logit[e]; i1 = e; }
    float w0 = 1.0f / (1.0f + expf(m1 - m0));   // renormalized top-2 softmax
    float w1 = 1.0f - w0;
    int s0 = atomicAdd(&counts[i0], 1);
    tids[i0 * ntok + s0] = n;  tws[i0 * ntok + s0] = w0;
    int s1 = atomicAdd(&counts[i1], 1);
    tids[i1 * ntok + s1] = n;  tws[i1 * ntok + s1] = w1;
  }
}

// One block = 32 tokens of one expert, 8 waves. bf16x3 split-precision GEMMs:
// h = gelu(W1t(A) @ x(B) + b1); y = W2t(A) @ h(B) + b2; atomic top-2 combine.
__global__ __launch_bounds__(256, 2)
void moe_ffn_kernel(const float* __restrict__ x,
                    const unsigned short* __restrict__ W1h,
                    const unsigned short* __restrict__ W1l,
                    const float* __restrict__ b1,
                    const unsigned short* __restrict__ W2h,
                    const unsigned short* __restrict__ W2l,
                    const float* __restrict__ b2,
                    const int* __restrict__ counts,
                    const int* __restrict__ tids,
                    const float* __restrict__ tws,
                    float* __restrict__ out,
                    int ntok) {
  extern __shared__ char smem[];
  __bf16* xh_s = (__bf16*)smem;                 // [ROWS][XB_STRIDE]
  __bf16* xl_s = xh_s + ROWS * XB_STRIDE;
  __bf16* hh_s = xl_s + ROWS * XB_STRIDE;       // [ROWS][HB_STRIDE]
  __bf16* hl_s = hh_s + ROWS * HB_STRIDE;
  int*   tid_s = (int*)(hl_s + ROWS * HB_STRIDE);
  float* w_s   = (float*)(tid_s + ROWS);

  const int e   = blockIdx.y;
  const int cnt = counts[e];
  const int r0  = blockIdx.x * ROWS;
  if (r0 >= cnt) return;
  const int m = min(ROWS, cnt - r0);

  const int t    = threadIdx.x;
  const int wv   = t >> 5;
  const int lane = t & 31;
  const int lm   = lane & 15;
  const int hi   = lane >> 4;
  const int hi8  = hi * 8;

  if (t < ROWS) {
    tid_s[t] = (t < m) ? tids[e * ntok + r0 + t] : -1;
    w_s[t]   = (t < m) ? tws[e * ntok + r0 + t] : 0.0f;
  }
  __syncthreads();

  // Stage x tile as bf16 hi/lo: thread -> (token row = t/8, 64-col chunk)
  {
    int row = t >> 3;
    int c0  = (t & 7) * 64;
    int tok = tid_s[row];
    uint2* dh = (uint2*)(xh_s + row * XB_STRIDE + c0);
    uint2* dl = (uint2*)(xl_s + row * XB_STRIDE + c0);
    if (tok >= 0) {
      const float4* src = (const float4*)(x + (size_t)tok * D_MODEL + c0);
      #pragma unroll
      for (int i = 0; i < 16; ++i) {
        float4 v = src[i];
        unsigned h0 = f2bf(v.x), h1 = f2bf(v.y), h2 = f2bf(v.z), h3 = f2bf(v.w);
        unsigned l0 = f2bf(v.x - bf2f(h0)), l1 = f2bf(v.y - bf2f(h1));
        unsigned l2 = f2bf(v.z - bf2f(h2)), l3 = f2bf(v.w - bf2f(h3));
        dh[i] = make_uint2(h0 | (h1 << 16), h2 | (h3 << 16));
        dl[i] = make_uint2(l0 | (l1 << 16), l2 | (l3 << 16));
      }
    } else {
      uint2 z = make_uint2(0u, 0u);
      #pragma unroll
      for (int i = 0; i < 16; ++i) { dh[i] = z; dl[i] = z; }
    }
  }
  __syncthreads();

  const float* b1e = b1 + e * D_FF;
  const float* b2e = b2 + e * D_MODEL;

  // GEMM1 tile for this wave: f rows [fT*16,+16) of chunk, tokens [nT*16,+16)
  const int fT = wv & 3;
  const int nT = wv >> 2;

  // GEMM2: wave owns d_out rows [wv*64, +64): 4 M-tiles x 2 N-tiles
  v8f yacc[4][2];
  #pragma unroll
  for (int ii = 0; ii < 4; ++ii)
    #pragma unroll
    for (int jj = 0; jj < 2; ++jj)
      yacc[ii][jj] = (v8f)(0.0f);

  for (int f0 = 0; f0 < D_FF; f0 += FCH) {
    // ---- GEMM1: h(16f x 16n) = W1t @ x, K = d_model, bf16x3 ----
    v8f hacc = (v8f)(0.0f);
    {
      size_t wrow = ((size_t)e * D_FF + f0 + fT * 16 + lm) * D_MODEL + hi8;
      const __bf16* ah = (const __bf16*)W1h + wrow;
      const __bf16* al = (const __bf16*)W1l + wrow;
      const __bf16* bh = xh_s + (nT * 16 + lm) * XB_STRIDE + hi8;
      const __bf16* bl = xl_s + (nT * 16 + lm) * XB_STRIDE + hi8;
      #pragma unroll 4
      for (int kc = 0; kc < D_MODEL; kc += 32) {
        v16bf Ah = ldA(ah + kc);
        v16bf Al = ldA(al + kc);
        v16bf Bh = ldB(bh + kc);
        v16bf Bl = ldB(bl + kc);
        hacc = wmma_bf(Ah, Bh, hacc);
        hacc = wmma_bf(Ah, Bl, hacc);
        hacc = wmma_bf(Al, Bh, hacc);
      }
    }
    __syncthreads();   // previous chunk's h readers done
    {
      // C layout: VGPR i -> f = f0 + fT*16 + hi8 + i ; lane column n = nT*16+lm
      const float4* bp = (const float4*)(b1e + f0 + fT * 16 + hi8);
      float4 bb0 = bp[0], bb1 = bp[1];
      float bias[8] = {bb0.x, bb0.y, bb0.z, bb0.w, bb1.x, bb1.y, bb1.z, bb1.w};
      v8u ph, pl;
      #pragma unroll
      for (int i = 0; i < 8; ++i) {
        float v = gelu_tanh(hacc[i] + bias[i]);
        unsigned short h = f2bf(v);
        ph[i] = h;
        pl[i] = f2bf(v - bf2f(h));
      }
      int n1 = nT * 16 + lm;
      *(v8u*)(hh_s + n1 * HB_STRIDE + fT * 16 + hi8) = ph;
      *(v8u*)(hl_s + n1 * HB_STRIDE + fT * 16 + hi8) = pl;
    }
    __syncthreads();   // h chunk ready

    // ---- GEMM2: y(64dout x 32n) += W2t @ h, K = f chunk, bf16x3 ----
    #pragma unroll
    for (int kc = 0; kc < FCH; kc += 32) {
      v16bf Bh[2], Bl[2];
      #pragma unroll
      for (int nt = 0; nt < 2; ++nt) {
        Bh[nt] = ldB(hh_s + (nt * 16 + lm) * HB_STRIDE + kc + hi8);
        Bl[nt] = ldB(hl_s + (nt * 16 + lm) * HB_STRIDE + kc + hi8);
      }
      #pragma unroll
      for (int mT = 0; mT < 4; ++mT) {
        size_t wrow = ((size_t)e * D_MODEL + wv * 64 + mT * 16 + lm) * D_FF
                      + f0 + kc + hi8;
        v16bf Ah = ldA((const __bf16*)W2h + wrow);
        v16bf Al = ldA((const __bf16*)W2l + wrow);
        #pragma unroll
        for (int nt = 0; nt < 2; ++nt) {
          yacc[mT][nt] = wmma_bf(Ah, Bh[nt], yacc[mT][nt]);
          yacc[mT][nt] = wmma_bf(Ah, Bl[nt], yacc[mT][nt]);
          yacc[mT][nt] = wmma_bf(Al, Bh[nt], yacc[mT][nt]);
        }
      }
    }
  }

  // ---- Epilogue: + b2, scale by routing weight, atomic top-2 combine ----
  #pragma unroll
  for (int mT = 0; mT < 4; ++mT) {
    #pragma unroll
    for (int nt = 0; nt < 2; ++nt) {
      int n = nt * 16 + lm;
      if (n < m) {
        int tok = tid_s[n];
        float wt = w_s[n];
        int dbase = wv * 64 + mT * 16 + hi8;
        const float* bp = b2e + dbase;
        float* op = out + (size_t)tok * D_MODEL + dbase;
        #pragma unroll
        for (int i = 0; i < 8; ++i)
          atomicAdd(op + i, (yacc[mT][nt][i] + bp[i]) * wt);
      }
    }
  }
}

extern "C" void kernel_launch(void* const* d_in, const int* in_sizes, int n_in,
                              void* d_out, int out_size, void* d_ws, size_t ws_size,
                              hipStream_t stream) {
  const float* x  = (const float*)d_in[0];
  const float* Wr = (const float*)d_in[1];
  const float* br = (const float*)d_in[2];
  const float* W1 = (const float*)d_in[3];
  const float* b1 = (const float*)d_in[4];
  const float* W2 = (const float*)d_in[5];
  const float* b2 = (const float*)d_in[6];
  float* out = (float*)d_out;

  int ntok = in_sizes[0] / D_MODEL;   // B*S tokens

  // workspace: counts | tids | tws | W1h | W1l | W2h | W2l
  char* ws = (char*)d_ws;
  int*   counts = (int*)ws;
  int*   tids   = (int*)(ws + 64);
  float* tws    = (float*)(ws + 64 + sizeof(int) * NEXP * ntok);
  size_t off = 64 + (size_t)8 * NEXP * ntok;
  off = (off + 255) & ~(size_t)255;
  const size_t wsz = (size_t)NEXP * D_MODEL * D_FF;   // elements per weight tensor
  unsigned short* W1hp = (unsigned short*)(ws + off);               off += 2 * wsz;
  unsigned short* W1lp = (unsigned short*)(ws + off);               off += 2 * wsz;
  unsigned short* W2hp = (unsigned short*)(ws + off);               off += 2 * wsz;
  unsigned short* W2lp = (unsigned short*)(ws + off);

  int nouts = ntok * D_MODEL;
  moe_zero_kernel<<<(nouts + 1023) / 1024, 256, 0, stream>>>(out, counts, nouts);

  // W1 [E][512][2048] -> [E][2048][512] bf16 hi/lo ; W2 [E][2048][512] -> [E][512][2048]
  transpose_split_kernel<<<dim3(D_FF / 32, D_MODEL / 32, NEXP), 256, 0, stream>>>(
      W1, W1hp, W1lp, D_MODEL, D_FF);
  transpose_split_kernel<<<dim3(D_MODEL / 32, D_FF / 32, NEXP), 256, 0, stream>>>(
      W2, W2hp, W2lp, D_FF, D_MODEL);

  moe_router_kernel<<<(ntok + 7) / 8, 256, 0, stream>>>(x, Wr, br, counts, tids, tws, ntok);

  dim3 grid((ntok + ROWS - 1) / ROWS, NEXP);
  size_t shbytes = (size_t)(2 * ROWS * XB_STRIDE + 2 * ROWS * HB_STRIDE) * sizeof(__bf16)
                 + (size_t)ROWS * 2 * sizeof(int);
  moe_ffn_kernel<<<grid, 256, shbytes, stream>>>(x, W1hp, W1lp, b1, W2hp, W2lp, b2,
                                                 counts, tids, tws, out, ntok);
}
